// MoHAttention_10574209483086
// MI455X (gfx1250) — matrix operand order repaired and checked by reading, compiled, not run on hardware
//
#include <hip/hip_runtime.h>
#include <hip/hip_fp16.h>

// ---------------------------------------------------------------------------
// MoH attention for gfx1250 (MI455X): wave32 + v_wmma_f32_16x16x32_f16
// B=32, S=512, D=1024, H=16, DK=64, H_SH=4, K_SEL=2, H_DYN=12
// ---------------------------------------------------------------------------

typedef __attribute__((ext_vector_type(16))) _Float16 v16h;
typedef __attribute__((ext_vector_type(8)))  float    v8f;

#define B_SZ   32
#define S_SZ   512
#define D_SZ   1024
#define H_SZ   16
#define DK_SZ  64
#define HDYN   12

union AFrag { v16h v; unsigned u[8]; };

__device__ inline v8f v8f_zero() {
  v8f z;
#pragma unroll
  for (int i = 0; i < 8; ++i) z[i] = 0.0f;
  return z;
}

__device__ inline v8f wmma_f16(const AFrag& a, const AFrag& b, v8f c) {
  // D = A(16x32 f16) * B(32x16 f16) + C(16x16 f32)
  return __builtin_amdgcn_wmma_f32_16x16x32_f16(
      /*neg_a=*/false, a.v, /*neg_b=*/false, b.v,
      /*c_mod=*/(short)0, c, /*reuse_a=*/false, /*reuse_b=*/false);
}

// ---------------------------------------------------------------------------
// Generic GEMM: out = A[M,K] * W[K,N] + bias,  f16 WMMA math, fp32 accum.
// A may be f32 (converted while staging to LDS) or f16.
// Epilogue can write f32, f16 row-major, and/or f16 transposed to [B,H,DK,S].
// Block: 256 threads (8 waves). Tile: 128(M) x 128(N), K-step 32.
// ---------------------------------------------------------------------------
__global__ __launch_bounds__(256)
void gemm_proj_kernel(const void* __restrict__ Aptr, int a_is_half,
                      const float* __restrict__ W, const float* __restrict__ bias,
                      __half* __restrict__ outh, float* __restrict__ outf,
                      __half* __restrict__ outtr,
                      int M, int N, int K) {
  __shared__ __half As[128 * 36];   // [m][k], stride 36 halves
  __shared__ __half BsT[128 * 36];  // [n][k], stride 36 halves

  const int tid  = threadIdx.x;
  const int lane = tid & 31;
  const int wave = tid >> 5;
  const int lo   = lane & 15;
  const int hi   = lane >> 4;
  const int m0   = blockIdx.y * 128;
  const int n0   = blockIdx.x * 128;
  const int mbase = wave * 16;

  v8f acc[8];
#pragma unroll
  for (int t = 0; t < 8; ++t) acc[t] = v8f_zero();

  for (int k0 = 0; k0 < K; k0 += 32) {
    // ---- stage A tile 128x32 into LDS (convert f32->f16 if needed) ----
    if (a_is_half) {
      const __half* A = (const __half*)Aptr;
#pragma unroll
      for (int p = 0; p < 4; ++p) {
        int row = p * 32 + (tid >> 3);
        int cg  = (tid & 7) * 4;
        const __half* src = A + (size_t)(m0 + row) * K + k0 + cg;
        if (k0 + 32 < K) __builtin_prefetch(src + 32, 0, 0);
        uint2 dv = *(const uint2*)src;
        *(uint2*)&As[row * 36 + cg] = dv;
      }
    } else {
      const float* A = (const float*)Aptr;
#pragma unroll
      for (int p = 0; p < 4; ++p) {
        int row = p * 32 + (tid >> 3);
        int cg  = (tid & 7) * 4;
        const float* src = A + (size_t)(m0 + row) * K + k0 + cg;
        if (k0 + 32 < K) __builtin_prefetch(src + 32, 0, 0);
        float4 f = *(const float4*)src;
        __half2 h01 = __floats2half2_rn(f.x, f.y);
        __half2 h23 = __floats2half2_rn(f.z, f.w);
        *(__half2*)&As[row * 36 + cg]     = h01;
        *(__half2*)&As[row * 36 + cg + 2] = h23;
      }
    }
    // ---- stage W tile 32x128 transposed into LDS as [n][k] ----
#pragma unroll
    for (int p = 0; p < 4; ++p) {
      int kr = p * 8 + (tid >> 5);
      int cg = (tid & 31) * 4;
      const float* src = W + (size_t)(k0 + kr) * N + n0 + cg;
      if (k0 + 32 < K) __builtin_prefetch(src + (size_t)32 * N, 0, 0);
      float4 f = *(const float4*)src;
      BsT[(cg + 0) * 36 + kr] = __float2half(f.x);
      BsT[(cg + 1) * 36 + kr] = __float2half(f.y);
      BsT[(cg + 2) * 36 + kr] = __float2half(f.z);
      BsT[(cg + 3) * 36 + kr] = __float2half(f.w);
    }
    __syncthreads();

    // ---- A fragment (this wave's 16 rows, K=32) ----
    AFrag af;
#pragma unroll
    for (int v = 0; v < 8; ++v) {
      int g = v >> 2, vv = v & 3;
      int kk = g * 16 + hi * 8 + vv * 2;
      af.u[v] = *(const unsigned*)&As[(mbase + lo) * 36 + kk];
    }
    // ---- 8 WMMAs across the 128-wide N tile, A reused ----
#pragma unroll
    for (int nt = 0; nt < 8; ++nt) {
      AFrag bf;
#pragma unroll
      for (int v = 0; v < 8; ++v) {
        int kk = (v >> 2) * 16 + hi * 8 + (v & 3) * 2;
        bf.u[v] = *(const unsigned*)&BsT[(nt * 16 + lo) * 36 + kk];
      }
      acc[nt] = wmma_f16(af, bf, acc[nt]);
    }
    __syncthreads();
  }

  // ---- epilogue: D layout: VGPR r -> M = r + 8*hi, N = lo ----
#pragma unroll
  for (int nt = 0; nt < 8; ++nt) {
#pragma unroll
    for (int r = 0; r < 8; ++r) {
      int gr = m0 + mbase + r + 8 * hi;
      int gc = n0 + nt * 16 + lo;
      float val = acc[nt][r] + bias[gc];
      if (outf)  outf[(size_t)gr * N + gc] = val;
      if (outh)  outh[(size_t)gr * N + gc] = __float2half(val);
      if (outtr) {
        int bb = gr >> 9, ss = gr & 511;   // row = b*S + s
        int hh = gc >> 6, dk = gc & 63;    // col = h*DK + dk
        outtr[(((size_t)bb * H_SZ + hh) * DK_SZ + dk) * S_SZ + ss] =
            __float2half(val);
      }
    }
  }
}

// ---------------------------------------------------------------------------
// Gate: per row i of qp [16384, 1024]: softmax(qp @ Wg) -> hard top-2 mask.
// Writes dyn [16384, 12] f32.
// ---------------------------------------------------------------------------
__global__ void gate_kernel(const __half* __restrict__ qp,
                            const float* __restrict__ Wg,
                            float* __restrict__ dyn) {
  int i = blockIdx.x * blockDim.x + threadIdx.x;
  if (i >= B_SZ * S_SZ) return;
  float acc[HDYN];
#pragma unroll
  for (int j = 0; j < HDYN; ++j) acc[j] = 0.0f;
  const __half* row = qp + (size_t)i * D_SZ;
  for (int d = 0; d < D_SZ; ++d) {
    float x = __half2float(row[d]);
    const float* wr = Wg + d * HDYN;
#pragma unroll
    for (int j = 0; j < HDYN; ++j) acc[j] += x * wr[j];
  }
  float mx = acc[0];
#pragma unroll
  for (int j = 1; j < HDYN; ++j) mx = fmaxf(mx, acc[j]);
  float g[HDYN], s = 0.0f;
#pragma unroll
  for (int j = 0; j < HDYN; ++j) { g[j] = __expf(acc[j] - mx); s += g[j]; }
  float inv = 1.0f / s;
#pragma unroll
  for (int j = 0; j < HDYN; ++j) g[j] *= inv;
  // top-2 (strict > keeps earliest index on ties, matching lax.top_k)
  int i0 = 0;
#pragma unroll
  for (int j = 1; j < HDYN; ++j) if (g[j] > g[i0]) i0 = j;
  int i1 = -1;
#pragma unroll
  for (int j = 0; j < HDYN; ++j)
    if (j != i0 && (i1 < 0 || g[j] > g[i1])) i1 = j;
#pragma unroll
  for (int j = 0; j < HDYN; ++j)
    dyn[(size_t)i * HDYN + j] = (j == i0 || j == i1) ? g[j] : 0.0f;
}

// ---------------------------------------------------------------------------
// routing[b][h]: h<4 -> 1.0 ; else mean over S of dyn[b,s,h-4]. Deterministic.
// ---------------------------------------------------------------------------
__global__ void routing_kernel(const float* __restrict__ dyn,
                               float* __restrict__ routing) {
  int t = blockIdx.x * blockDim.x + threadIdx.x;
  if (t >= B_SZ * H_SZ) return;
  int b = t / H_SZ, h = t % H_SZ;
  float r;
  if (h < 4) {
    r = 1.0f;
  } else {
    float s = 0.0f;
    for (int ss = 0; ss < S_SZ; ++ss)
      s += dyn[((size_t)(b * S_SZ + ss)) * HDYN + (h - 4)];
    r = s * (1.0f / (float)S_SZ);
  }
  routing[t] = r;
}

// ---------------------------------------------------------------------------
// Flash attention. Block = 256 threads (8 waves), one block per
// (b, h, 128 q-rows). Each wave: 16 q rows, online softmax over 32-key
// chunks, PV via WMMA with V pre-transposed to [b,h,dk,s].
// Row 0 of scores is forced to 0 (reference quirk). Routing folded in.
// ---------------------------------------------------------------------------
__global__ __launch_bounds__(256)
void flash_kernel(const __half* __restrict__ qp, const __half* __restrict__ kp,
                  const __half* __restrict__ vT,
                  const float* __restrict__ routing,
                  __half* __restrict__ ctx) {
  __shared__ __half Pl[8][16 * 36];  // per-wave P tile [16 rows][32 keys]

  const int tid  = threadIdx.x;
  const int lane = tid & 31;
  const int wave = tid >> 5;
  const int lo   = lane & 15;
  const int hi   = lane >> 4;

  const int qt = blockIdx.x & 3;       // S/128 = 4 q tiles
  const int bh = blockIdx.x >> 2;      // 0..511
  const int b  = bh >> 4;
  const int h  = bh & 15;
  const int qrow = qt * 128 + wave * 16;
  const float rscale = routing[bh];

  // Q A-fragments for feature halves [0,32) and [32,64)
  AFrag qa[2];
  const __half* qbase = qp + (size_t)(b * S_SZ + qrow + lo) * D_SZ + h * DK_SZ;
#pragma unroll
  for (int t = 0; t < 2; ++t)
#pragma unroll
    for (int v = 0; v < 8; ++v) {
      int g = v >> 2, vv = v & 3;
      int kk = t * 32 + g * 16 + hi * 8 + vv * 2;
      qa[t].u[v] = *(const unsigned*)(qbase + kk);
    }

  v8f accc[4];
#pragma unroll
  for (int t = 0; t < 4; ++t) accc[t] = v8f_zero();
  float rm[8], rl[8];
#pragma unroll
  for (int r = 0; r < 8; ++r) { rm[r] = -3.0e38f; rl[r] = 0.0f; }

  const __half* vb = vT + (size_t)bh * DK_SZ * S_SZ;  // [64][512]

  for (int c = 0; c < 16; ++c) {
    const int kbase = c * 32;
    v8f s0 = v8f_zero(), s1 = v8f_zero();
    // scores: S[m][n] = sum_f Q[m,f] * K[n,f]
#pragma unroll
    for (int t = 0; t < 2; ++t) {
      const __half* kb =
          kp + (size_t)(b * S_SZ + kbase + t * 16 + lo) * D_SZ + h * DK_SZ;
#pragma unroll
      for (int step = 0; step < 2; ++step) {
        AFrag bf;
#pragma unroll
        for (int v = 0; v < 8; ++v)
          bf.u[v] = *(const unsigned*)(kb + step * 32 + (v >> 2) * 16 + hi * 8 + (v & 3) * 2);
        if (t == 0) s0 = wmma_f16(qa[step], bf, s0);
        else        s1 = wmma_f16(qa[step], bf, s1);
      }
    }

    // mask + online softmax; stash exp(P) into per-wave LDS slice
#pragma unroll
    for (int r = 0; r < 8; ++r) {
      int m  = r + 8 * hi;
      int qg = qrow + m;
      float v0 = s0[r] * 0.125f;           // 1/sqrt(64)
      float v1 = s1[r] * 0.125f;
      if (kbase + lo      > qg) v0 = -1.0e9f;
      if (kbase + 16 + lo > qg) v1 = -1.0e9f;
      if (qg == 0) { v0 = 0.0f; v1 = 0.0f; }  // reference: row 0 scores = 0
      float mx = fmaxf(v0, v1);
#pragma unroll
      for (int off = 1; off < 16; off <<= 1) mx = fmaxf(mx, __shfl_xor(mx, off));
      float newm = fmaxf(rm[r], mx);
      float sc = __expf(rm[r] - newm);
      float p0 = __expf(v0 - newm);
      float p1 = __expf(v1 - newm);
      float ps = p0 + p1;
#pragma unroll
      for (int off = 1; off < 16; off <<= 1) ps += __shfl_xor(ps, off);
      rl[r] = rl[r] * sc + ps;
      rm[r] = newm;
#pragma unroll
      for (int t = 0; t < 4; ++t) accc[t][r] *= sc;
      Pl[wave][m * 36 + lo]      = __float2half(p0);
      Pl[wave][m * 36 + 16 + lo] = __float2half(p1);
    }
    __syncthreads();  // uniform across all 8 waves every chunk

    // P A-fragment (16 rows x 32 keys) from LDS
    AFrag pa;
#pragma unroll
    for (int v = 0; v < 8; ++v) {
      int g = v >> 2, vv = v & 3;
      int kk = g * 16 + hi * 8 + vv * 2;
      pa.u[v] = *(const unsigned*)&Pl[wave][lo * 36 + kk];
    }
    // ctx += P * V : B[k][n] = V[kbase+k][ft*16+n] from transposed vT
#pragma unroll
    for (int ft = 0; ft < 4; ++ft) {
      AFrag bf;
      const __half* col = vb + (size_t)(ft * 16 + lo) * S_SZ + kbase;
#pragma unroll
      for (int v = 0; v < 8; ++v)
        bf.u[v] = *(const unsigned*)(col + (v >> 2) * 16 + hi * 8 + (v & 3) * 2);
      accc[ft] = wmma_f16(pa, bf, accc[ft]);
    }
  }

  // epilogue: normalize, apply routing, write ctx as [B,S,H*DK] f16
#pragma unroll
  for (int ft = 0; ft < 4; ++ft)
#pragma unroll
    for (int r = 0; r < 8; ++r) {
      int m  = r + 8 * hi;
      int qg = qrow + m;
      float val = accc[ft][r] / rl[r] * rscale;
      ctx[(size_t)(b * S_SZ + qg) * D_SZ + h * DK_SZ + ft * 16 + lo] =
          __float2half(val);
    }
}

// ---------------------------------------------------------------------------
extern "C" void kernel_launch(void* const* d_in, const int* in_sizes, int n_in,
                              void* d_out, int out_size, void* d_ws,
                              size_t ws_size, hipStream_t stream) {
  (void)in_sizes; (void)n_in; (void)out_size; (void)ws_size;
  const float* q  = (const float*)d_in[0];
  const float* k  = (const float*)d_in[1];
  const float* v  = (const float*)d_in[2];
  /* d_in[3] = mask: causal tril, recomputed analytically */
  const float* Wq = (const float*)d_in[4];
  const float* bq = (const float*)d_in[5];
  const float* Wk = (const float*)d_in[6];
  const float* bk = (const float*)d_in[7];
  const float* Wv = (const float*)d_in[8];
  const float* bv = (const float*)d_in[9];
  const float* Wg = (const float*)d_in[10];
  const float* Wo = (const float*)d_in[11];
  const float* bo = (const float*)d_in[12];
  float* out = (float*)d_out;

  const size_t HALF_MAT = (size_t)B_SZ * S_SZ * D_SZ * sizeof(__half); // 32 MiB
  char* ws = (char*)d_ws;
  __half* qp_h    = (__half*)(ws);
  __half* kp_h    = (__half*)(ws + HALF_MAT);
  __half* vT      = (__half*)(ws + 2 * HALF_MAT);
  __half* ctxh    = (__half*)(ws + 3 * HALF_MAT);
  float*  dyn     = (float*)(ws + 4 * HALF_MAT);
  float*  routing = (float*)(ws + 4 * HALF_MAT + (size_t)B_SZ * S_SZ * HDYN * 4);

  const int M = B_SZ * S_SZ;  // 16384
  dim3 blk(256);
  dim3 ggrid(D_SZ / 128, M / 128);  // 8 x 128

  gemm_proj_kernel<<<ggrid, blk, 0, stream>>>(q, 0, Wq, bq, qp_h, nullptr,
                                              nullptr, M, D_SZ, D_SZ);
  gemm_proj_kernel<<<ggrid, blk, 0, stream>>>(k, 0, Wk, bk, kp_h, nullptr,
                                              nullptr, M, D_SZ, D_SZ);
  gemm_proj_kernel<<<ggrid, blk, 0, stream>>>(v, 0, Wv, bv, nullptr, nullptr,
                                              vT, M, D_SZ, D_SZ);

  gate_kernel<<<(M + 255) / 256, 256, 0, stream>>>(qp_h, Wg, dyn);
  routing_kernel<<<(B_SZ * H_SZ + 255) / 256, 256, 0, stream>>>(dyn, routing);

  flash_kernel<<<B_SZ * H_SZ * (S_SZ / 128), 256, 0, stream>>>(
      qp_h, kp_h, vT, routing, ctxh);

  gemm_proj_kernel<<<ggrid, blk, 0, stream>>>(ctxh, 1, Wo, bo, nullptr, out,
                                              nullptr, M, D_SZ, D_SZ);
}
